// CNN4d_78297253806219
// MI455X (gfx1250) — matrix operand order repaired and checked by reading, not compiled, not run on hardware
//
#include <hip/hip_runtime.h>
#include <stdint.h>

typedef __attribute__((ext_vector_type(16))) __bf16 v16bf;
typedef __attribute__((ext_vector_type(8)))  float  v8f;

#define NPOS 262144            // 2*8*16*32*32 output positions
#define NROWS 8192             // 2*8*16*32 (b,t,z,y) rows of 32 x-positions
#define XP   40                // padded x extent: 2 left halo, 32 valid, 6 right halo

static __device__ __forceinline__ int imax(int a, int b) { return a > b ? a : b; }
static __device__ __forceinline__ int imin(int a, int b) { return a < b ? a : b; }

// ---------------------------------------------------------------------------
// Weight pre-pack: f32 W[Cout][Cin][5][5][5][5] -> bf16 fragments laid out in
// per-lane A-matrix order (contiguous K = 16*(lane/16) + i), so the conv
// kernel loads one contiguous v16bf (32B) per lane per K-chunk.
// CIN==16: chunk = (dt,dz,dy, dx-pair {0,1},{2,3},{4,phantom5(zero)}) -> 375
// CIN==32: chunk = (dt,dz,dy,dx) -> 625
// wpack index = ((mt*nchunk + chunk)*32 + lane)*16 + i
// ---------------------------------------------------------------------------
template<int CIN>
__global__ void prep_weights(const float* __restrict__ W, __bf16* __restrict__ wp,
                             int nchunk, int mts) {
    int id = blockIdx.x * blockDim.x + threadIdx.x;
    int total = mts * nchunk * 512;
    if (id >= total) return;
    int i     = id & 15;
    int lane  = (id >> 4) & 31;
    int chunk = (id >> 9) % nchunk;
    int mt    = (id >> 9) / nchunk;
    int m = lane & 15;
    // contiguous per-lane K mapping: K = 16*(lane/16) + i (matches B fragment reads)
    int K = (lane >> 4) * 16 + i;
    int cout = mt * 16 + m;
    int dt, dz, dy, dx, cin;
    bool zero = false;
    if (CIN == 16) {
        int tg = chunk / 3, ck = chunk % 3;
        dx  = 2 * ck + (K >> 4);
        cin = K & 15;
        if (dx > 4) { zero = true; dx = 0; }
        dy = tg % 5; dz = (tg / 5) % 5; dt = tg / 25;
    } else {
        dx = chunk % 5;
        int tg = chunk / 5;
        dy = tg % 5; dz = (tg / 5) % 5; dt = tg / 25;
        cin = K;
    }
    float w = zero ? 0.0f
        : W[(((((size_t)cout * CIN + cin) * 5 + dt) * 5 + dz) * 5 + dy) * 5 + dx];
    wp[id] = (__bf16)w;
}

// ---------------------------------------------------------------------------
__global__ void zero_u32(uint32_t* p, int n) {
    int i = blockIdx.x * blockDim.x + threadIdx.x;
    if (i < n) p[i] = 0u;
}

// ---------------------------------------------------------------------------
// Embedding: x[pos][8] @ w_emb[8][16] + b_emb -> bf16 act (channels-last, x-padded)
// ---------------------------------------------------------------------------
__global__ void embed_kernel(const float* __restrict__ x, const float* __restrict__ wemb,
                             const float* __restrict__ bemb, __bf16* __restrict__ act) {
    int p = blockIdx.x * blockDim.x + threadIdx.x;
    if (p >= NPOS) return;
    float xf[8];
#pragma unroll
    for (int f = 0; f < 8; ++f) xf[f] = x[(size_t)p * 8 + f];
    int row = p >> 5, xx = p & 31;
    __bf16* o = act + ((size_t)row * XP + xx + 2) * 16;
#pragma unroll
    for (int c = 0; c < 16; ++c) {
        float s = bemb[c];
#pragma unroll
        for (int f = 0; f < 8; ++f) s += xf[f] * wemb[f * 16 + c];
        o[c] = (__bf16)s;
    }
}

// ---------------------------------------------------------------------------
// Implicit-GEMM 4D conv via v_wmma_f32_16x16x32_bf16.
// One wave = one full output row (32 x = two N=16 tiles) x (MT*16 Cout).
// Tile id is readfirstlane'd; tap boundaries handled by CLAMPED scalar loop
// bounds -> the inner body is straight-line (no phi copies of accumulators,
// no WMMA->VALU hazard NOPs), EXEC stays all-1s for WMMA.
// Activation row (XP x CIN bf16) is staged once per tap group into
// wave-private LDS with global_load_async_to_lds_* (ASYNCcnt), then B
// fragments come from LDS; A fragments stream from the packed weights.
// ---------------------------------------------------------------------------
template<int CIN, int MT>
__global__ void __launch_bounds__(256)
conv4d_wmma(const __bf16* __restrict__ act, const __bf16* __restrict__ wpack,
            const float* __restrict__ bias, float* __restrict__ out) {
    const int CPW    = (CIN == 16) ? 3 : 5;     // K-chunks per (dt,dz,dy) tap group
    const int NCHUNK = (CIN == 16) ? 375 : 625;
    const int COUT   = MT * 16;
    const int ROWE   = XP * CIN;                // row elements (640 / 1280)
    const int NFULL  = (ROWE * 2) / 512;        // full 512B async b128 steps (2 / 5)

    __shared__ __bf16 smem[8][1280];            // 2560B per wave, 20KB per block

    int wid  = threadIdx.x >> 5;
    int lane = threadIdx.x & 31;
    int nlo  = lane & 15;
    int half = lane >> 4;

    // scalarize the tile id: all loop bounds below become SGPR values
    int tile = __builtin_amdgcn_readfirstlane(blockIdx.x * 8 + wid);
    int y = tile & 31, z = (tile >> 5) & 15, t = (tile >> 9) & 7, b = (tile >> 12) & 1;

    // clamped tap ranges: tt=t+dt-2 in [0,8), zz in [0,16), yy in [0,32)
    int dtlo = imax(0, 2 - t), dthi = imin(5, 10 - t);
    int dzlo = imax(0, 2 - z), dzhi = imin(5, 18 - z);
    int dylo = imax(0, 2 - y), dyhi = imin(5, 34 - y);

    const __bf16* sp = &smem[wid][0];
    uint32_t ldsbase = (uint32_t)(uintptr_t)sp;   // low 32 bits of generic ptr = LDS offset

    v8f acc00 = (v8f){}, acc01 = (v8f){};
    v8f acc10 = (v8f){}, acc11 = (v8f){};

    for (int dt = dtlo; dt < dthi; ++dt) {
        int tt = t + dt - 2;
        for (int dz = dzlo; dz < dzhi; ++dz) {
            int zz = z + dz - 2;
            for (int dy = dylo; dy < dyhi; ++dy) {
                int yy = y + dy - 2;

                int row = (((b * 8 + tt) * 16 + zz) * 32) + yy;
                const __bf16* rowp = act + (size_t)row * ROWE;

                // ---- async stage: one padded activation row -> this wave's LDS
                {
                    uint64_t g = (uint64_t)(uintptr_t)rowp + (uint32_t)(lane * 16);
                    uint32_t l = ldsbase + (uint32_t)(lane * 16);
#pragma unroll
                    for (int s = 0; s < NFULL; ++s) {
                        asm volatile("global_load_async_to_lds_b128 %0, %1, off offset:%2"
                                     :: "v"(l), "v"(g), "i"(s * 512) : "memory");
                    }
                    if (CIN == 16) {    // 256B tail
                        uint64_t g2 = (uint64_t)(uintptr_t)rowp + (uint32_t)(lane * 8);
                        uint32_t l2 = ldsbase + (uint32_t)(lane * 8);
                        asm volatile("global_load_async_to_lds_b64 %0, %1, off offset:1024"
                                     :: "v"(l2), "v"(g2) : "memory");
                    }
                    asm volatile("s_wait_asynccnt 0x0" ::: "memory");
                }

                int tg = (dt * 5 + dz) * 5 + dy;
                const __bf16* wpb = wpack + (size_t)(tg * CPW) * 512 + lane * 16;
                __builtin_prefetch(wpb + (size_t)CPW * 512, 0, 3);
#pragma unroll
                for (int ck = 0; ck < CPW; ++ck) {
                    int dx   = (CIN == 16) ? (2 * ck + half) : ck;
                    int kofs = (CIN == 32) ? (half * 16) : 0;
                    v16bf a0 = *(const v16bf*)(wpb + ck * 512);
                    v16bf a1 = a0;
                    if (MT == 2)
                        a1 = *(const v16bf*)(wpb + ck * 512 + (size_t)NCHUNK * 512);

                    v16bf bf0 = *(const v16bf*)(sp + (nlo + dx) * CIN + kofs);
                    v16bf bf1 = *(const v16bf*)(sp + (16 + nlo + dx) * CIN + kofs);

                    acc00 = __builtin_amdgcn_wmma_f32_16x16x32_bf16(
                        false, a0, false, bf0, (short)0, acc00, false, false);
                    acc10 = __builtin_amdgcn_wmma_f32_16x16x32_bf16(
                        false, a0, false, bf1, (short)0, acc10, false, false);
                    if (MT == 2) {
                        acc01 = __builtin_amdgcn_wmma_f32_16x16x32_bf16(
                            false, a1, false, bf0, (short)0, acc01, false, false);
                        acc11 = __builtin_amdgcn_wmma_f32_16x16x32_bf16(
                            false, a1, false, bf1, (short)0, acc11, false, false);
                    }
                }
            }
        }
    }

    // C/D layout: VGPR r, lanes 0-15 -> M=r, lanes 16-31 -> M=r+8; N = lane&15
    int mbase = half * 8;
    {
        v8f a = acc00;
#pragma unroll
        for (int r = 0; r < 8; ++r) a[r] += bias[mbase + r];
        float* op = out + (size_t)(tile * 32 + nlo) * COUT + mbase;
        *(float4*)(op)     = make_float4(a[0], a[1], a[2], a[3]);
        *(float4*)(op + 4) = make_float4(a[4], a[5], a[6], a[7]);
    }
    {
        v8f a = acc10;
#pragma unroll
        for (int r = 0; r < 8; ++r) a[r] += bias[mbase + r];
        float* op = out + (size_t)(tile * 32 + 16 + nlo) * COUT + mbase;
        *(float4*)(op)     = make_float4(a[0], a[1], a[2], a[3]);
        *(float4*)(op + 4) = make_float4(a[4], a[5], a[6], a[7]);
    }
    if (MT == 2) {
        {
            v8f a = acc01;
#pragma unroll
            for (int r = 0; r < 8; ++r) a[r] += bias[16 + mbase + r];
            float* op = out + (size_t)(tile * 32 + nlo) * COUT + 16 + mbase;
            *(float4*)(op)     = make_float4(a[0], a[1], a[2], a[3]);
            *(float4*)(op + 4) = make_float4(a[4], a[5], a[6], a[7]);
        }
        {
            v8f a = acc11;
#pragma unroll
            for (int r = 0; r < 8; ++r) a[r] += bias[16 + mbase + r];
            float* op = out + (size_t)(tile * 32 + 16 + nlo) * COUT + 16 + mbase;
            *(float4*)(op)     = make_float4(a[0], a[1], a[2], a[3]);
            *(float4*)(op + 4) = make_float4(a[4], a[5], a[6], a[7]);
        }
    }
}

// ---------------------------------------------------------------------------
// BatchNorm stats: one block per channel, deterministic tree reduction.
// stats[2c] = mean, stats[2c+1] = rsqrt(var + eps)
// ---------------------------------------------------------------------------
__global__ void bn_stats(const float* __restrict__ cv, float* __restrict__ stats, int C) {
    int c = blockIdx.x;
    float s = 0.f, q = 0.f;
    for (int p = threadIdx.x; p < NPOS; p += blockDim.x) {
        float v = cv[(size_t)p * C + c];
        s += v; q += v * v;
    }
    __shared__ float ss[256], qq[256];
    ss[threadIdx.x] = s; qq[threadIdx.x] = q;
    __syncthreads();
    for (int st = 128; st > 0; st >>= 1) {
        if (threadIdx.x < st) {
            ss[threadIdx.x] += ss[threadIdx.x + st];
            qq[threadIdx.x] += qq[threadIdx.x + st];
        }
        __syncthreads();
    }
    if (threadIdx.x == 0) {
        float m   = ss[0] / (float)NPOS;
        float var = qq[0] / (float)NPOS - m * m;
        stats[2 * c]     = m;
        stats[2 * c + 1] = rsqrtf(var + 1e-5f);
    }
}

// ---------------------------------------------------------------------------
// BN + leaky-ReLU, repack to bf16 channels-last x-padded activation buffer.
// ---------------------------------------------------------------------------
__global__ void bn_apply(const float* __restrict__ cv, const float* __restrict__ stats,
                         const float* __restrict__ g, const float* __restrict__ be,
                         __bf16* __restrict__ act, int C) {
    int id = blockIdx.x * blockDim.x + threadIdx.x;
    if (id >= NPOS * C) return;
    int c = id % C;
    int p = id / C;
    float m = stats[2 * c], inv = stats[2 * c + 1];
    float v = cv[id];
    float yv = (v - m) * inv * g[c] + be[c];
    yv = (yv > 0.f) ? yv : 0.01f * yv;
    int row = p >> 5, xx = p & 31;
    act[((size_t)row * XP + xx + 2) * C + c] = (__bf16)yv;
}

// ---------------------------------------------------------------------------
// Final BN + leaky-ReLU + H->1 projection, f32 straight to d_out.
// ---------------------------------------------------------------------------
__global__ void bn_proj(const float* __restrict__ cv, const float* __restrict__ stats,
                        const float* __restrict__ g, const float* __restrict__ be,
                        const float* __restrict__ wproj, const float* __restrict__ bproj,
                        float* __restrict__ out) {
    int p = blockIdx.x * blockDim.x + threadIdx.x;
    if (p >= NPOS) return;
    float acc = bproj[0];
#pragma unroll
    for (int c = 0; c < 16; ++c) {
        float v = cv[(size_t)p * 16 + c];
        float yv = (v - stats[2 * c]) * stats[2 * c + 1] * g[c] + be[c];
        yv = (yv > 0.f) ? yv : 0.01f * yv;
        acc += yv * wproj[c];
    }
    out[p] = acc;
}

// ---------------------------------------------------------------------------
// Host launcher
// ---------------------------------------------------------------------------
extern "C" void kernel_launch(void* const* d_in, const int* in_sizes, int n_in,
                              void* d_out, int out_size, void* d_ws, size_t ws_size,
                              hipStream_t stream) {
    const float* x      = (const float*)d_in[0];
    const float* w_emb  = (const float*)d_in[1];
    const float* b_emb  = (const float*)d_in[2];
    const float* W1     = (const float*)d_in[3];
    const float* b1     = (const float*)d_in[4];
    const float* g1     = (const float*)d_in[5];
    const float* beta1  = (const float*)d_in[6];
    const float* W2     = (const float*)d_in[7];
    const float* b2     = (const float*)d_in[8];
    const float* g2     = (const float*)d_in[9];
    const float* beta2  = (const float*)d_in[10];
    const float* W3     = (const float*)d_in[11];
    const float* b3     = (const float*)d_in[12];
    const float* g3     = (const float*)d_in[13];
    const float* beta3  = (const float*)d_in[14];
    const float* w_proj = (const float*)d_in[15];
    const float* b_proj = (const float*)d_in[16];
    float* out = (float*)d_out;

    char* ws = (char*)d_ws;
    const size_t OFF_WP1  = 0;                                // 375*512*2    = 384,000 B
    const size_t OFF_WP2  = OFF_WP1 + 384000;                 // 750*512*2    = 768,000 B
    const size_t OFF_WP3  = OFF_WP2 + 768000;                 // 625*512*2    = 640,000 B
    const size_t OFF_ACTA = OFF_WP3 + 640000;                 // 8192*40*32*2 = 20,971,520 B
    const size_t OFF_ACTB = OFF_ACTA + 20971520;              // 20,971,520 B
    const size_t OFF_CONV = OFF_ACTB + 20971520;              // 262144*32*4  = 33,554,432 B
    const size_t OFF_STAT = OFF_CONV + 33554432;              // 64*4 B

    __bf16* wp1  = (__bf16*)(ws + OFF_WP1);
    __bf16* wp2  = (__bf16*)(ws + OFF_WP2);
    __bf16* wp3  = (__bf16*)(ws + OFF_WP3);
    __bf16* actA = (__bf16*)(ws + OFF_ACTA);
    __bf16* actB = (__bf16*)(ws + OFF_ACTB);
    float*  conv = (float*)(ws + OFF_CONV);
    float*  stat = (float*)(ws + OFF_STAT);

    // Weight pre-pack into WMMA A-fragment order
    prep_weights<16><<<(1 * 375 * 512 + 255) / 256, 256, 0, stream>>>(W1, wp1, 375, 1);
    prep_weights<16><<<(2 * 375 * 512 + 255) / 256, 256, 0, stream>>>(W2, wp2, 375, 2);
    prep_weights<32><<<(1 * 625 * 512 + 255) / 256, 256, 0, stream>>>(W3, wp3, 625, 1);

    // Zero both padded activation buffers (establishes zero halos)
    {
        int n32 = (int)((20971520u * 2u) / 4u);
        zero_u32<<<(n32 + 255) / 256, 256, 0, stream>>>((uint32_t*)actA, n32);
    }

    // Embedding -> actA (C=16)
    embed_kernel<<<NPOS / 256, 256, 0, stream>>>(x, w_emb, b_emb, actA);

    // Layer 1: conv(16->16), BN stats, BN+leaky -> actB (C=16)
    conv4d_wmma<16, 1><<<NROWS / 8, 256, 0, stream>>>(actA, wp1, b1, conv);
    bn_stats<<<16, 256, 0, stream>>>(conv, stat, 16);
    bn_apply<<<(NPOS * 16) / 256, 256, 0, stream>>>(conv, stat, g1, beta1, actB, 16);

    // Layer 2: conv(16->32), BN stats, BN+leaky -> actA (C=32)
    conv4d_wmma<16, 2><<<NROWS / 8, 256, 0, stream>>>(actB, wp2, b2, conv);
    bn_stats<<<32, 256, 0, stream>>>(conv, stat, 32);
    bn_apply<<<(NPOS * 32) / 256, 256, 0, stream>>>(conv, stat, g2, beta2, actA, 32);

    // Layer 3: conv(32->16), BN stats, fused BN+leaky+projection -> out
    conv4d_wmma<32, 1><<<NROWS / 8, 256, 0, stream>>>(actA, wp3, b3, conv);
    bn_stats<<<16, 256, 0, stream>>>(conv, stat, 16);
    bn_proj<<<NPOS / 256, 256, 0, stream>>>(conv, stat, g3, beta3, w_proj, b_proj, out);
}
